// OptimizedMoELayer_22565758173959
// MI455X (gfx1250) — compile-verified, matrix operand order checked by reading, not run on hardware
//
#include <hip/hip_runtime.h>
#include <hip/hip_bf16.h>
#include <math.h>

#define NUM_EXPERTS 8
#define TOPK 2
#define HIDDEN 512
#define INTER 2048
#define NTOK 2048                        /* B*S */
#define NSLOT (NTOK * TOPK)              /* 4096 */
#define MTILE 64                         /* rows per GEMM tile; expert segs 64-aligned */
#define RPAD (NSLOT + NUM_EXPERTS * MTILE) /* 4608 padded rows */
#define NTILES (RPAD / MTILE)            /* 72 tiles of 64 rows */

typedef __bf16 bf16;
typedef __attribute__((ext_vector_type(16))) __bf16 v16bf;
typedef __attribute__((ext_vector_type(8)))  __bf16 v8bf;
typedef __attribute__((ext_vector_type(8)))  float  v8f;
typedef __attribute__((ext_vector_type(4)))  float  v4f;

// ---------------------------------------------------------------- utilities

// jax.nn.gelu (approximate=True), branchless tanh via exp
__device__ __forceinline__ float gelu_tanh(float x) {
    float u = 0.7978845608028654f * (x + 0.044715f * x * x * x);
    float t = __expf(-2.0f * fabsf(u));
    float th = __fdividef(1.0f - t, 1.0f + t);
    th = copysignf(th, u);
    return 0.5f * x * (1.0f + th);
}

// pack two floats into one dword of bf16 (lowers to v_cvt_pk_bf16_f32)
__device__ __forceinline__ unsigned int pkbf(float a, float b) {
    unsigned short lo = __builtin_bit_cast(unsigned short, (bf16)a);
    unsigned short hi = __builtin_bit_cast(unsigned short, (bf16)b);
    return ((unsigned int)hi << 16) | (unsigned int)lo;
}

// ---------------------------------------------------------------- kernel 0: zero control block

__global__ void moe_zero_ctrl(int* counts) {
    if (threadIdx.x < NUM_EXPERTS) counts[threadIdx.x] = 0;
}

// ---------------------------------------------------------------- kernel 1: router

__global__ void moe_router(const float* __restrict__ hs,
                           const float* __restrict__ wr,
                           const float* __restrict__ br,
                           float* __restrict__ probs,
                           int*   __restrict__ top_idx,
                           float* __restrict__ wflat,
                           int*   __restrict__ counts) {
    int t = blockIdx.x * blockDim.x + threadIdx.x;
    if (t >= NTOK) return;
    const float* x = hs + (size_t)t * HIDDEN;

    float acc[NUM_EXPERTS];
#pragma unroll
    for (int e = 0; e < NUM_EXPERTS; ++e) acc[e] = br[e];
    for (int h = 0; h < HIDDEN; ++h) {
        float xv = x[h];
        const float* wrow = wr + h * NUM_EXPERTS;
#pragma unroll
        for (int e = 0; e < NUM_EXPERTS; ++e) acc[e] += xv * wrow[e];
    }
    // softmax
    float m = acc[0];
#pragma unroll
    for (int e = 1; e < NUM_EXPERTS; ++e) m = fmaxf(m, acc[e]);
    float p[NUM_EXPERTS];
    float s = 0.f;
#pragma unroll
    for (int e = 0; e < NUM_EXPERTS; ++e) { p[e] = expf(acc[e] - m); s += p[e]; }
    float inv = 1.0f / s;
#pragma unroll
    for (int e = 0; e < NUM_EXPERTS; ++e) { p[e] *= inv; probs[t * NUM_EXPERTS + e] = p[e]; }
    // top-2, lax.top_k tie rule: first index wins
    int i0 = 0;
#pragma unroll
    for (int e = 1; e < NUM_EXPERTS; ++e) if (p[e] > p[i0]) i0 = e;
    int i1 = (i0 == 0) ? 1 : 0;
#pragma unroll
    for (int e = 0; e < NUM_EXPERTS; ++e) if (e != i0 && e != i1 && p[e] > p[i1]) i1 = e;
    float v0 = p[i0], v1 = p[i1], rs = 1.0f / (v0 + v1);
    top_idx[t * 2 + 0] = i0;  wflat[t * 2 + 0] = v0 * rs;
    top_idx[t * 2 + 1] = i1;  wflat[t * 2 + 1] = v1 * rs;
    atomicAdd(&counts[i0], 1);
    atomicAdd(&counts[i1], 1);
}

// ---------------------------------------------------------------- kernel 2: balance loss (deterministic reduce)

__global__ void moe_balance(const float* __restrict__ probs,
                            const int*   __restrict__ counts,
                            float* __restrict__ loss_out) {
    __shared__ float red[256];
    int tid = threadIdx.x;
    float part[NUM_EXPERTS];
#pragma unroll
    for (int e = 0; e < NUM_EXPERTS; ++e) part[e] = 0.f;
    for (int t = tid; t < NTOK; t += 256) {
#pragma unroll
        for (int e = 0; e < NUM_EXPERTS; ++e) part[e] += probs[t * NUM_EXPERTS + e];
    }
    float loss = 0.f;
    for (int e = 0; e < NUM_EXPERTS; ++e) {
        red[tid] = part[e];
        __syncthreads();
        for (int s = 128; s > 0; s >>= 1) {
            if (tid < s) red[tid] += red[tid + s];
            __syncthreads();
        }
        if (tid == 0) {
            float P = red[0] / (float)NTOK;
            float f = (float)counts[e] / (float)NSLOT;
            loss += f * P;
        }
        __syncthreads();
    }
    if (tid == 0) loss_out[0] = 0.01f * (float)NUM_EXPERTS * loss;
}

// ---------------------------------------------------------------- kernel 3: segment scan + tile map (64-aligned segments)

__global__ void moe_scan(const int* __restrict__ counts,
                         int* __restrict__ cursor,
                         int* __restrict__ offsets,
                         int* __restrict__ tile_e) {
    if (threadIdx.x != 0 || blockIdx.x != 0) return;
    int off = 0, tile = 0;
    for (int e = 0; e < NUM_EXPERTS; ++e) {
        offsets[e] = off;
        cursor[e]  = off;
        int cap = (counts[e] + MTILE - 1) & ~(MTILE - 1);
        int nt = cap / MTILE;
        for (int i = 0; i < nt; ++i) tile_e[tile++] = e;
        off += cap;
    }
    for (; tile < NTILES; ++tile) tile_e[tile] = 0;
}

// ---------------------------------------------------------------- kernel 4: scatter slots + gather rows (f32 -> bf16, packed stores)

__global__ void moe_scatter(const float* __restrict__ hs,
                            const int*   __restrict__ top_idx,
                            int* __restrict__ cursor,
                            int* __restrict__ slot_pos,
                            bf16* __restrict__ Abuf) {
    int s = blockIdx.x * blockDim.x + threadIdx.x;
    if (s >= NSLOT) return;
    int e = top_idx[s];
    int pos = atomicAdd(&cursor[e], 1);
    slot_pos[s] = pos;
    const float* src = hs + (size_t)(s >> 1) * HIDDEN;
    bf16* dst = Abuf + (size_t)pos * HIDDEN;
    for (int h = 0; h < HIDDEN; h += 4) {
        v4f v = *(const v4f*)(src + h);
        uint2 packed;
        packed.x = pkbf(v[0], v[1]);
        packed.y = pkbf(v[2], v[3]);
        *(uint2*)(dst + h) = packed;
    }
}

// ---------------------------------------------------------------- WMMA GEMM
// Block = 256 threads = 8 waves. Block tile M=64, N=128; K staged 64 per barrier.
// Wave w handles m-subtile (w&3)*16 and n-half (w>>2)*64 -> 4 accumulators,
// 8 v_wmma per barrier pair (2 k-steps x 4 n-groups), A fragments reused 4x.
// A in LDS [64][64]; B in LDS transposed [128][64] so fragments are ds_load_b128.

__device__ __forceinline__ v16bf ld_frag(const bf16* base, int roff) {
    v8bf lo = *(const v8bf*)(base + roff);
    v8bf hi = *(const v8bf*)(base + roff + 16);
    v16bf f;
#pragma unroll
    for (int j = 0; j < 8; ++j) { f[j] = lo[j]; f[j + 8] = hi[j]; }
    return f;
}

template <int KDIM, int LDB, bool DO_GELU>
__global__ __launch_bounds__(256)
void moe_gemm(const bf16*  __restrict__ A,       // [RPAD][KDIM]
              const float* __restrict__ W,       // [E][KDIM][LDB]
              const float* __restrict__ bias,    // [E][LDB]
              bf16*  __restrict__ outB,          // bf16 out (GELU path), stride LDB
              float* __restrict__ outF,          // f32 out, stride LDB
              const int* __restrict__ tile_e) {
    __shared__ bf16 sA[64 * 64];     // [m][k]
    __shared__ bf16 sB[128 * 64];    // [n][k]
    const int tid  = threadIdx.x, wave = tid >> 5, lane = tid & 31;
    const int tile = blockIdx.y,  n0   = blockIdx.x * 128;
    const int e = tile_e[tile], row0 = tile * MTILE;
    const float* B = W + (size_t)e * KDIM * LDB;

    const int mi = (wave & 3) * 16;            // wave's m-subtile base
    const int nh = (wave >> 2) * 64;           // wave's n-half base
    const int m  = mi + (lane & 15);
    const int kb = (lane >> 4) * 8;
    const int nn = nh + (lane & 15);

    v8f acc[4] = {{}, {}, {}, {}};
    for (int k0 = 0; k0 < KDIM; k0 += 64) {
        // --- stage A tile: 64x64 bf16, two 16B chunks per thread ---
#pragma unroll
        for (int i = 0; i < 2; ++i) {
            int idx = tid * 8 + 2048 * i;
            int am = idx >> 6, ak = idx & 63;
            const bf16* gp = A + (size_t)(row0 + am) * KDIM + k0 + ak;
            *(v8bf*)(&sA[am * 64 + ak]) = *(const v8bf*)gp;
        }
        // --- stage B tile transposed: 64x128 f32 -> bf16 [n][k], packed b32 stores ---
        // groups (k-pair x n-octet): 32 x 16 = 512 groups, 2 per thread
#pragma unroll
        for (int i = 0; i < 2; ++i) {
            int g  = tid + 256 * i;
            int kp = g >> 4;             // 0..31 -> k = 2*kp
            int no = g & 15;             // 0..15 -> n = 8*no
            int k  = 2 * kp;
            int n  = 8 * no;
            const float* r0 = B + (size_t)(k0 + k) * LDB + n0 + n;
            const float* r1 = r0 + LDB;
            v4f a0 = ((const v4f*)r0)[0];
            v4f a1 = ((const v4f*)r0)[1];
            v4f b0 = ((const v4f*)r1)[0];
            v4f b1 = ((const v4f*)r1)[1];
            if (i == 0 && k0 + 64 < KDIM)
                __builtin_prefetch(r0 + (size_t)64 * LDB, 0, 1);
            unsigned int* dst = (unsigned int*)&sB[0];
#pragma unroll
            for (int j = 0; j < 4; ++j) {
                dst[((n + j) * 64 + k) >> 1]     = pkbf(a0[j], b0[j]);
                dst[((n + 4 + j) * 64 + k) >> 1] = pkbf(a1[j], b1[j]);
            }
        }
        __syncthreads();
        // --- 8 WMMAs per stage: 2 k-steps x 4 n-groups, A fragment reused ---
#pragma unroll
        for (int s = 0; s < 2; ++s) {
            v16bf af = ld_frag(sA, m * 64 + kb + s * 32);
#pragma unroll
            for (int gg = 0; gg < 4; ++gg) {
                v16bf bf = ld_frag(sB, (nn + gg * 16) * 64 + kb + s * 32);
                acc[gg] = __builtin_amdgcn_wmma_f32_16x16x32_bf16(
                    false, af, false, bf, (short)0, acc[gg], false, false);
            }
        }
        __syncthreads();
    }
    // --- epilogue ---
#pragma unroll
    for (int gg = 0; gg < 4; ++gg) {
        const int ncol = n0 + nn + gg * 16;
        const float bv = bias[e * LDB + ncol];
#pragma unroll
        for (int r = 0; r < 8; ++r) {
            int mrow = mi + r + 8 * (lane >> 4);
            float x = acc[gg][r] + bv;
            if (DO_GELU) {
                outB[(size_t)(row0 + mrow) * LDB + ncol] = (bf16)gelu_tanh(x);
            } else {
                outF[(size_t)(row0 + mrow) * LDB + ncol] = x;
            }
        }
    }
}

// ---------------------------------------------------------------- kernel 7: combine + residual (vectorized x4)

__global__ void moe_combine(const float* __restrict__ hs,
                            const float* __restrict__ Y,
                            const int*   __restrict__ slot_pos,
                            const float* __restrict__ wflat,
                            float* __restrict__ out) {
    int idx4 = blockIdx.x * 256 + threadIdx.x;    // < NTOK*HIDDEN/4
    int idx = idx4 * 4;
    int t = idx >> 9, h = idx & (HIDDEN - 1);
    int s0 = t * 2, s1 = s0 + 1;
    float w0 = wflat[s0], w1 = wflat[s1];
    const float* y0 = Y + (size_t)slot_pos[s0] * HIDDEN + h;
    const float* y1 = Y + (size_t)slot_pos[s1] * HIDDEN + h;
    v4f r  = *(const v4f*)(hs + idx);
    v4f a  = *(const v4f*)y0;
    v4f b  = *(const v4f*)y1;
#pragma unroll
    for (int j = 0; j < 4; ++j) r[j] += w0 * a[j] + w1 * b[j];
    *(v4f*)(out + idx) = r;
}

// ---------------------------------------------------------------- launch

extern "C" void kernel_launch(void* const* d_in, const int* in_sizes, int n_in,
                              void* d_out, int out_size, void* d_ws, size_t ws_size,
                              hipStream_t stream) {
    (void)in_sizes; (void)n_in; (void)out_size; (void)ws_size;
    const float* hs = (const float*)d_in[0];   // (2,1024,512)
    const float* wr = (const float*)d_in[1];   // (512,8)
    const float* br = (const float*)d_in[2];   // (8,)
    const float* w1 = (const float*)d_in[3];   // (8,512,2048)
    const float* b1 = (const float*)d_in[4];   // (8,2048)
    const float* w2 = (const float*)d_in[5];   // (8,2048,512)
    const float* b2 = (const float*)d_in[6];   // (8,512)
    float* out = (float*)d_out;                // 2*1024*512 floats + 1 loss scalar

    char* ws = (char*)d_ws;
    int*   counts   = (int*)(ws + 0);
    int*   cursor   = counts + 8;
    int*   offsets  = cursor + 8;
    int*   tile_e   = offsets + 8;                        // NTILES ints (<4KB total ctrl)
    float* probs    = (float*)(ws + 4096);                // NTOK*8   (64 KB)
    int*   top_idx  = (int*)  (ws + 69632);               // NSLOT    (16 KB)
    float* wflat    = (float*)(ws + 86016);               // NSLOT    (16 KB)
    int*   slot_pos = (int*)  (ws + 102400);              // NSLOT    (16 KB)
    bf16*  Abuf     = (bf16*) (ws + 118784);              // RPAD*512 bf16  (~4.7 MB)
    bf16*  Hmid     = (bf16*) (ws + 4837376);             // RPAD*2048 bf16 (~18.9 MB)
    float* Ybuf     = (float*)(ws + 23711744);            // RPAD*512 f32   (~9.4 MB)
    float* loss_out = out + (size_t)NTOK * HIDDEN;

    moe_zero_ctrl<<<1, 32, 0, stream>>>(counts);
    moe_router<<<NTOK / 256, 256, 0, stream>>>(hs, wr, br, probs, top_idx, wflat, counts);
    moe_balance<<<1, 256, 0, stream>>>(probs, counts, loss_out);
    moe_scan<<<1, 32, 0, stream>>>(counts, cursor, offsets, tile_e);
    moe_scatter<<<NSLOT / 256, 256, 0, stream>>>(hs, top_idx, cursor, slot_pos, Abuf);

    dim3 g1(INTER / 128, NTILES);
    moe_gemm<HIDDEN, INTER, true><<<g1, 256, 0, stream>>>(Abuf, w1, b1, Hmid, nullptr, tile_e);
    dim3 g2(HIDDEN / 128, NTILES);
    moe_gemm<INTER, HIDDEN, false><<<g2, 256, 0, stream>>>(Hmid, w2, b2, nullptr, Ybuf, tile_e);

    moe_combine<<<(NTOK * HIDDEN) / (256 * 4), 256, 0, stream>>>(hs, Ybuf, slot_pos, wflat, out);
}
